// AttnBlockpp_15599321219621
// MI455X (gfx1250) — matrix operand order repaired and checked by reading, not compiled, not run on hardware
//
#include <hip/hip_runtime.h>
#include <hip/hip_bf16.h>
#include <math.h>
#include <stdint.h>

// ---------------------------------------------------------------------------
// AttnBlock (GroupNorm -> q,k,v 1x1 conv -> flash attention -> proj -> +x)
// B=4, C=128, H=W=64 (N=4096 tokens), 32 groups.
// Matmuls via v_wmma_f32_16x16x32_f16; K/V staging via
// global_load_async_to_lds_b128 (ASYNCcnt) with double buffering.
// ---------------------------------------------------------------------------

typedef __attribute__((ext_vector_type(16))) _Float16 v16h;
typedef __attribute__((ext_vector_type(8)))  _Float16 v8h;
typedef __attribute__((ext_vector_type(8)))  float    v8f;

#define C_DIM 128
#define N_TOK 4096
#define B_DIM 4

__device__ __forceinline__ v8f wmma_f16(v16h a, v16h b, v8f c) {
  return __builtin_amdgcn_wmma_f32_16x16x32_f16(false, a, false, b, (short)0, c,
                                                false, false);
}

// Async global->LDS copy, 16 bytes per lane. LDS address = low 32 bits of the
// generic shared pointer (LDS aperture: addr[31:0]).
__device__ __forceinline__ void async_ld_b128(void* lds, const void* gptr) {
  asm volatile("global_load_async_to_lds_b128 %0, %1, off"
               :: "v"((uint32_t)(uintptr_t)lds), "v"(gptr)
               : "memory");
}
__device__ __forceinline__ void wait_async0() {
  asm volatile("s_wait_asynccnt 0" ::: "memory");
}

// A fragment (16x32 f16, MxK), row-major source, row stride in halves.
// lane m = lane&15; halves 0..15 -> K = hi*16+0..15.
__device__ __forceinline__ v16h load_afrag(const _Float16* base, int row0,
                                           int kbase, int stride, int lane) {
  int m = lane & 15, hi = lane >> 4;
  const _Float16* p = base + (size_t)(row0 + m) * stride + kbase;
  return *(const v16h*)(p + hi * 16);
}

// B fragment (32x16 f16, KxN), source laid out [n][k], row stride in halves.
// Lanes 0-15: n=lane, K=0..15 ; lanes 16-31: n=lane-16, K=16..31.
__device__ __forceinline__ v16h load_bfrag(const _Float16* base, int n0,
                                           int kbase, int stride, int lane) {
  int n = lane & 15, hi = lane >> 4;
  const _Float16* p = base + (size_t)(n0 + n) * stride + kbase + hi * 16;
  return *(const v16h*)p;
}

// ---------------------------------------------------------------------------
// 1) GroupNorm statistics: block per (b, group); group = 4 contiguous
// channels * 4096 = 16384 contiguous floats.
// ---------------------------------------------------------------------------
__global__ __launch_bounds__(256) void gn_stats_kernel(const float* __restrict__ x,
                                                       float* __restrict__ stats) {
  __shared__ float rs[256], rq[256];
  int bg = blockIdx.x;
  const float* base = x + (size_t)bg * 16384;
  float s = 0.f, q = 0.f;
  for (int i = threadIdx.x; i < 16384; i += 256) {
    float v = base[i];
    s += v; q += v * v;
  }
  int tid = threadIdx.x;
  rs[tid] = s; rq[tid] = q;
  __syncthreads();
  for (int off = 128; off > 0; off >>= 1) {
    if (tid < off) { rs[tid] += rs[tid + off]; rq[tid] += rq[tid + off]; }
    __syncthreads();
  }
  if (tid == 0) {
    float mean = rs[0] * (1.f / 16384.f);
    float var = rq[0] * (1.f / 16384.f) - mean * mean;
    stats[bg * 2] = mean;
    stats[bg * 2 + 1] = rsqrtf(var + 1e-6f);
  }
}

// ---------------------------------------------------------------------------
// 2) Apply GroupNorm, write token-major f16 hT[b][n][c].
// ---------------------------------------------------------------------------
__global__ __launch_bounds__(256) void gn_apply_kernel(const float* __restrict__ x,
                                                       const float* __restrict__ stats,
                                                       const float* __restrict__ gamma,
                                                       const float* __restrict__ beta,
                                                       _Float16* __restrict__ hT) {
  size_t i = (size_t)blockIdx.x * 256 + threadIdx.x;  // total 4*128*4096
  int n = (int)(i & 4095);
  int c = (int)((i >> 12) & 127);
  int b = (int)(i >> 19);
  int g = c >> 2;
  float mean = stats[(b * 32 + g) * 2];
  float rstd = stats[(b * 32 + g) * 2 + 1];
  float h = (x[i] - mean) * rstd * gamma[c] + beta[c];
  hT[((size_t)b * N_TOK + n) * C_DIM + c] = (_Float16)h;
}

// ---------------------------------------------------------------------------
// 3) 1x1 conv via WMMA: out[cout][n] = sum_cin W[cin][cout] * hT[n][cin] + b.
// One wave per 16(cout) x 16(token) tile, 8 waves/block sharing a W tile in
// LDS. mode 0: token-major f16; mode 1: channel-major f16; mode 2: f32
// residual out = x + D + bias.
// ---------------------------------------------------------------------------
__global__ __launch_bounds__(256) void nin_kernel(const _Float16* __restrict__ hT,
                                                  const float* __restrict__ W,
                                                  const float* __restrict__ bias,
                                                  _Float16* __restrict__ outH,
                                                  float* __restrict__ outF,
                                                  const float* __restrict__ xres,
                                                  int mode) {
  __shared__ __align__(32) _Float16 wl[16 * 128];  // [cout16][cin128]
  int wave = threadIdx.x >> 5, lane = threadIdx.x & 31;
  int tile = blockIdx.x * 8 + wave;          // 8192 tiles total
  int ntile = tile & 255;                    // same mtile/b across a block
  int mtile = (tile >> 8) & 7;
  int b = tile >> 11;
  int n0 = ntile * 16, m0 = mtile * 16;

  for (int i = threadIdx.x; i < 2048; i += 256) {
    int mo = i >> 7, ci = i & 127;
    wl[mo * 128 + ci] = (_Float16)W[ci * 128 + m0 + mo];
  }
  __syncthreads();

  const _Float16* hb = hT + (size_t)b * N_TOK * C_DIM;
  v8f acc = {};
#pragma unroll
  for (int cc = 0; cc < 4; cc++) {
    v16h a = load_afrag(wl, 0, cc * 32, 128, lane);
    v16h bf = load_bfrag(hb, n0, cc * 32, 128, lane);
    acc = wmma_f16(a, bf, acc);
  }

  int m = lane & 15, hi = lane >> 4;
  int tok = n0 + m;
#pragma unroll
  for (int r = 0; r < 8; r++) {
    int cout = m0 + r + hi * 8;
    float val = acc[r] + bias[cout];
    if (mode == 0) {
      outH[((size_t)b * N_TOK + tok) * C_DIM + cout] = (_Float16)val;
    } else if (mode == 1) {
      outH[((size_t)b * C_DIM + cout) * N_TOK + tok] = (_Float16)val;
    } else {
      size_t off = ((size_t)b * C_DIM + cout) * N_TOK + tok;
      outF[off] = xres[off] + val;
    }
  }
}

// ---------------------------------------------------------------------------
// 4) Flash attention. Each wave: 16 queries, O(16x128) f32 accumulator.
// Block (8 waves) double-buffers 32-key K/V tiles in LDS via async
// global->LDS DMA. S = Q K^T * C^-1/2, online softmax, O += P V^T.
// ---------------------------------------------------------------------------
__global__ __launch_bounds__(256) void attn_kernel(const _Float16* __restrict__ qT,
                                                   const _Float16* __restrict__ kT,
                                                   const _Float16* __restrict__ vC,
                                                   _Float16* __restrict__ oT) {
  __shared__ __align__(32) _Float16 kt[2][32 * 128];   // [key32][ch128]
  __shared__ __align__(32) _Float16 vt[2][128 * 32];   // [ch128][key32]
  __shared__ __align__(32) _Float16 pl[8][16 * 32];    // per-wave P tile

  int wave = threadIdx.x >> 5, lane = threadIdx.x & 31;
  int tid = threadIdx.x;
  int b = blockIdx.x >> 5;
  int qblk = blockIdx.x & 31;
  int q0 = qblk * 128 + wave * 16;
  int m = lane & 15, hi = lane >> 4;

  // per-thread staging coordinates (16 halves each for K and V tiles)
  const _Float16* kgb = kT + ((size_t)b * N_TOK + (tid >> 3)) * C_DIM + (tid & 7) * 16;
  const _Float16* vgb = vC + ((size_t)b * C_DIM + (tid >> 1)) * N_TOK + (tid & 1) * 16;
  const int klo = (tid >> 3) * 128 + (tid & 7) * 16;
  const int vlo = (tid >> 1) * 32 + (tid & 1) * 16;

  // issue async stage of tile 0
  async_ld_b128(&kt[0][klo],     kgb);
  async_ld_b128(&kt[0][klo + 8], kgb + 8);
  async_ld_b128(&vt[0][vlo],     vgb);
  async_ld_b128(&vt[0][vlo + 8], vgb + 8);

  const _Float16* qb = qT + (size_t)b * N_TOK * C_DIM;
  v16h qa[4];
#pragma unroll
  for (int cc = 0; cc < 4; cc++) qa[cc] = load_afrag(qb, q0, cc * 32, 128, lane);

  v8f o[8];
#pragma unroll
  for (int cc = 0; cc < 8; cc++) o[cc] = (v8f){};
  float mrow[8], lrow[8];
#pragma unroll
  for (int r = 0; r < 8; r++) { mrow[r] = -INFINITY; lrow[r] = 0.f; }

  const float scale = 0.0883883476483f;  // 128^-0.5
  _Float16* pw = &pl[wave][0];

#pragma unroll 1
  for (int it = 0; it < N_TOK / 32; ++it) {
    wait_async0();      // this wave's async writes (tile `it`) done
    __syncthreads();    // whole tile visible; prev-iter reads of next buf done

    const _Float16* kc = &kt[it & 1][0];
    const _Float16* vc = &vt[it & 1][0];

    if (it + 1 < N_TOK / 32) {  // prefetch tile it+1 into the other buffer
      int nb = (it + 1) & 1;
      const _Float16* kg = kgb + (size_t)(it + 1) * 32 * C_DIM;
      const _Float16* vg = vgb + (it + 1) * 32;
      async_ld_b128(&kt[nb][klo],     kg);
      async_ld_b128(&kt[nb][klo + 8], kg + 8);
      async_ld_b128(&vt[nb][vlo],     vg);
      async_ld_b128(&vt[nb][vlo + 8], vg + 8);
    }

    v8f s0 = {}, s1 = {};
#pragma unroll
    for (int cc = 0; cc < 4; cc++) {
      v16h b0 = load_bfrag(kc, 0, cc * 32, 128, lane);
      v16h b1 = load_bfrag(kc, 16, cc * 32, 128, lane);
      s0 = wmma_f16(qa[cc], b0, s0);
      s1 = wmma_f16(qa[cc], b1, s1);
    }

    float p0[8], p1[8], alpha[8];
#pragma unroll
    for (int r = 0; r < 8; r++) {
      float a0 = s0[r] * scale, a1 = s1[r] * scale;
      float rm = fmaxf(a0, a1);
      rm = fmaxf(rm, __shfl_xor(rm, 1, 32));
      rm = fmaxf(rm, __shfl_xor(rm, 2, 32));
      rm = fmaxf(rm, __shfl_xor(rm, 4, 32));
      rm = fmaxf(rm, __shfl_xor(rm, 8, 32));
      float mn = fmaxf(mrow[r], rm);
      alpha[r] = __expf(mrow[r] - mn);
      mrow[r] = mn;
      p0[r] = __expf(a0 - mn);
      p1[r] = __expf(a1 - mn);
      float rs = p0[r] + p1[r];
      rs += __shfl_xor(rs, 1, 32);
      rs += __shfl_xor(rs, 2, 32);
      rs += __shfl_xor(rs, 4, 32);
      rs += __shfl_xor(rs, 8, 32);
      lrow[r] = lrow[r] * alpha[r] + rs;
      // stash P (C/D layout: row = r + 8*hi, col = lane&15)
      pw[(r + hi * 8) * 32 + m] = (_Float16)p0[r];
      pw[(r + hi * 8) * 32 + 16 + m] = (_Float16)p1[r];
    }

    // rescale running O by alpha (rows match accumulator layout)
#pragma unroll
    for (int cc = 0; cc < 8; cc++)
#pragma unroll
      for (int r = 0; r < 8; r++) o[cc][r] *= alpha[r];

    // reload P as A fragment (16 queries x 32 keys), O += P * V^T
    v16h pa = load_afrag(pw, 0, 0, 32, lane);
#pragma unroll
    for (int cc = 0; cc < 8; cc++) {
      v16h vb = load_bfrag(vc, cc * 16, 0, 32, lane);
      o[cc] = wmma_f16(pa, vb, o[cc]);
    }
  }

  // epilogue: normalize by row sums, store token-major f16 attn output
  _Float16* ob = oT + (size_t)b * N_TOK * C_DIM;
#pragma unroll
  for (int cc = 0; cc < 8; cc++)
#pragma unroll
    for (int r = 0; r < 8; r++) {
      float val = o[cc][r] / lrow[r];
      ob[(size_t)(q0 + r + hi * 8) * C_DIM + cc * 16 + m] = (_Float16)val;
    }
}

// ---------------------------------------------------------------------------
// Launch
// ---------------------------------------------------------------------------
extern "C" void kernel_launch(void* const* d_in, const int* in_sizes, int n_in,
                              void* d_out, int out_size, void* d_ws, size_t ws_size,
                              hipStream_t stream) {
  const float* x = (const float*)d_in[0];
  const float* gn_scale = (const float*)d_in[1];
  const float* gn_bias = (const float*)d_in[2];
  const float* W0 = (const float*)d_in[3];
  const float* b0 = (const float*)d_in[4];
  const float* W1 = (const float*)d_in[5];
  const float* b1 = (const float*)d_in[6];
  const float* W2 = (const float*)d_in[7];
  const float* b2 = (const float*)d_in[8];
  const float* W3 = (const float*)d_in[9];
  const float* b3 = (const float*)d_in[10];
  float* out = (float*)d_out;

  // workspace layout (~21 MB)
  char* wsb = (char*)d_ws;
  const size_t SZ = (size_t)B_DIM * N_TOK * C_DIM * sizeof(_Float16);  // 4 MiB
  float* stats = (float*)wsb;                       // 1 KiB
  _Float16* hT = (_Float16*)(wsb + 4096);           // [b][n][c]
  _Float16* qT = (_Float16*)(wsb + 4096 + SZ);      // [b][n][c]
  _Float16* kT = (_Float16*)(wsb + 4096 + 2 * SZ);  // [b][n][c]
  _Float16* vC = (_Float16*)(wsb + 4096 + 3 * SZ);  // [b][c][n]
  _Float16* aT = (_Float16*)(wsb + 4096 + 4 * SZ);  // [b][n][c] attn out

  gn_stats_kernel<<<B_DIM * 32, 256, 0, stream>>>(x, stats);
  gn_apply_kernel<<<(B_DIM * C_DIM * N_TOK) / 256, 256, 0, stream>>>(
      x, stats, gn_scale, gn_bias, hT);

  // q, k token-major; v channel-major
  nin_kernel<<<1024, 256, 0, stream>>>(hT, W0, b0, qT, nullptr, nullptr, 0);
  nin_kernel<<<1024, 256, 0, stream>>>(hT, W1, b1, kT, nullptr, nullptr, 0);
  nin_kernel<<<1024, 256, 0, stream>>>(hT, W2, b2, vC, nullptr, nullptr, 1);

  attn_kernel<<<B_DIM * 32, 256, 0, stream>>>(qT, kT, vC, aT);

  // final projection + residual
  nin_kernel<<<1024, 256, 0, stream>>>(aT, W3, b3, nullptr, out, x, 2);
}
